// T2SBlock_66881230734018
// MI455X (gfx1250) — matrix operand order repaired and checked by reading, not compiled, not run on hardware
//
#include <hip/hip_runtime.h>
#include <math.h>

// ---------------------------------------------------------------------------
// Shapes (fixed by the reference)
// ---------------------------------------------------------------------------
#define BATCH 64
#define DMODEL 1024
#define NHEAD 16
#define HDIM 64
#define TCTX 2048
#define DFF_ 4096

typedef __attribute__((ext_vector_type(16))) _Float16 v16h;
typedef __attribute__((ext_vector_type(8)))  _Float16 v8h;
typedef __attribute__((ext_vector_type(8)))  float    v8f;

#define BK 32
#define APAD 40   // LDS row stride in halves (80 B) -> bank-spread, 16B aligned

// ---------------------------------------------------------------------------
// GEMM: out[64 x N] = A[64 x K] @ W[K x N] + bias (+ optional relu / residual)
//
// Block = 256 threads (8 waves), owns 64 rows x 64 cols at ntile0.
// Per K-step of 32: all threads cooperatively stage A(64x32) and B(32x64)
// from global (coalesced float4) into LDS as f16 (A row-major, B transposed),
// then wave w computes m-tile (w&3) x n-tiles {2*(w>>2), 2*(w>>2)+1} with two
// v_wmma_f32_16x16x32_f16, fragments pulled from LDS as contiguous
// ds_load_b128s. 2 accumulators per wave -> no register spills. Every weight
// element is read from HBM exactly once.
// ---------------------------------------------------------------------------
__global__ void gemm64_wmma(const float* __restrict__ A,
                            const float* __restrict__ W,
                            const float* __restrict__ bias,
                            const float* __restrict__ resid,
                            float* __restrict__ out,
                            int K, int N, int relu) {
    const int tid    = threadIdx.x;
    const int lane   = tid & 31;
    const int wave   = tid >> 5;            // 0..7
    const int ntile0 = blockIdx.x * 64;

    const int mt  = wave & 3;               // m-tile 0..3
    const int ntp = (wave >> 2) * 2;        // first of two n-tiles

    __shared__ _Float16 aS[64][APAD];       // 5 KB   a[m][k]
    __shared__ _Float16 bT[64][APAD];       // 5 KB   b[col][k] (transposed)

    // staging coordinates
    const int am = tid >> 2;                // A: row 0..63
    const int ak = (tid & 3) * 8;           // A: 8 consecutive k
    const int bk = tid >> 3;                // B: k-row 0..31
    const int bc = (tid & 7) * 8;           // B: 8 consecutive cols

    v8f c0 = {}, c1 = {};

    for (int k0 = 0; k0 < K; k0 += BK) {
        // ---- stage A slice: 64 x 32 fp32 -> f16 LDS --------------------
        {
            const float4* p4 = reinterpret_cast<const float4*>(
                A + (size_t)am * K + k0 + ak);
            const float4 f0 = p4[0], f1 = p4[1];
            v8h p;
            p[0] = (_Float16)f0.x; p[1] = (_Float16)f0.y;
            p[2] = (_Float16)f0.z; p[3] = (_Float16)f0.w;
            p[4] = (_Float16)f1.x; p[5] = (_Float16)f1.y;
            p[6] = (_Float16)f1.z; p[7] = (_Float16)f1.w;
            *reinterpret_cast<v8h*>(&aS[am][ak]) = p;
        }
        // ---- stage B slice: 32 x 64 fp32 -> f16 LDS, transposed --------
        {
            const float4* p4 = reinterpret_cast<const float4*>(
                W + (size_t)(k0 + bk) * N + ntile0 + bc);
            const float4 f0 = p4[0], f1 = p4[1];
            bT[bc + 0][bk] = (_Float16)f0.x;
            bT[bc + 1][bk] = (_Float16)f0.y;
            bT[bc + 2][bk] = (_Float16)f0.z;
            bT[bc + 3][bk] = (_Float16)f0.w;
            bT[bc + 4][bk] = (_Float16)f1.x;
            bT[bc + 5][bk] = (_Float16)f1.y;
            bT[bc + 6][bk] = (_Float16)f1.z;
            bT[bc + 7][bk] = (_Float16)f1.w;
        }
        __syncthreads();

        // ---- fragments from LDS ---------------------------------------
        // A frag: lanes 0-15 m=lane (K 0-15); lanes 16-31 m=lane-16
        //         (K 16-31).
        const int m    = mt * 16 + (lane & 15);
        const int ksel = (lane >> 4) * 16;
        const v8h alo = *reinterpret_cast<const v8h*>(&aS[m][ksel]);
        const v8h ahi = *reinterpret_cast<const v8h*>(&aS[m][ksel + 8]);
        const v16h a = __builtin_shufflevector(alo, ahi,
            0, 1, 2, 3, 4, 5, 6, 7, 8, 9, 10, 11, 12, 13, 14, 15);

        // B frag: lanes 0-15 col n, K 0-15; lanes 16-31 col n, K 16-31.
        const int kh = (lane >> 4) * 16;
        const int n0 = (ntp + 0) * 16 + (lane & 15);
        const int n1 = (ntp + 1) * 16 + (lane & 15);
        const v8h b0lo = *reinterpret_cast<const v8h*>(&bT[n0][kh]);
        const v8h b0hi = *reinterpret_cast<const v8h*>(&bT[n0][kh + 8]);
        const v8h b1lo = *reinterpret_cast<const v8h*>(&bT[n1][kh]);
        const v8h b1hi = *reinterpret_cast<const v8h*>(&bT[n1][kh + 8]);
        const v16h b0 = __builtin_shufflevector(b0lo, b0hi,
            0, 1, 2, 3, 4, 5, 6, 7, 8, 9, 10, 11, 12, 13, 14, 15);
        const v16h b1 = __builtin_shufflevector(b1lo, b1hi,
            0, 1, 2, 3, 4, 5, 6, 7, 8, 9, 10, 11, 12, 13, 14, 15);

        c0 = __builtin_amdgcn_wmma_f32_16x16x32_f16(false, a, false, b0,
                                                    (short)0, c0, false, false);
        c1 = __builtin_amdgcn_wmma_f32_16x16x32_f16(false, a, false, b1,
                                                    (short)0, c1, false, false);
        __syncthreads();
    }

    // ---- epilogue: D layout VGPR v -> row v + 8*(lane>=16), col lane&15 ---
    const int nlo  = lane & 15;
    const int mofs = (lane >> 4) * 8;
#pragma unroll
    for (int v = 0; v < 8; ++v) {
        const int row = mt * 16 + mofs + v;
        const size_t ro = (size_t)row * N;
#pragma unroll
        for (int t = 0; t < 2; ++t) {
            const int col = ntile0 + (ntp + t) * 16 + nlo;
            float val = (t == 0 ? c0[v] : c1[v]) + bias[col];
            if (relu)  val = fmaxf(val, 0.0f);
            if (resid) val += resid[ro + col];
            out[ro + col] = val;
        }
    }
}

// ---------------------------------------------------------------------------
// Attention: one block per (b, h). Scores over t in [0, len]; t == len uses the
// freshly-projected k/v (we must NOT mutate the input caches). Two-pass softmax
// with scores staged in LDS, then p @ V with a 4-way t-chunk reduction.
// ---------------------------------------------------------------------------
__global__ void attn_kernel(const float* __restrict__ qkv,      // [B, 3*D]
                            const float* __restrict__ k_cache,  // [B,H,T,HD]
                            const float* __restrict__ v_cache,  // [B,H,T,HD]
                            const int*   __restrict__ kv_len,   // [B]
                            float* __restrict__ out) {          // [B, D]
    const int b   = blockIdx.x;
    const int h   = blockIdx.y;
    const int tid = threadIdx.x;   // 0..255

    __shared__ __align__(16) float q_s[HDIM];
    __shared__ __align__(16) float kn_s[HDIM];
    __shared__ __align__(16) float vn_s[HDIM];
    __shared__ float sc[TCTX];
    __shared__ float red[8];
    __shared__ float bc[2];
    __shared__ float oacc[4][HDIM];

    const size_t qoff = (size_t)b * (3 * DMODEL) + h * HDIM;
    if (tid < HDIM) {
        q_s[tid]  = qkv[qoff + tid];
        kn_s[tid] = qkv[qoff + DMODEL + tid];
        vn_s[tid] = qkv[qoff + 2 * DMODEL + tid];
    }
    __syncthreads();

    const int len = kv_len[b];                       // 1024 .. 2046
    const float* kbase = k_cache + (((size_t)b * NHEAD + h) * TCTX) * HDIM;
    const float* vbase = v_cache + (((size_t)b * NHEAD + h) * TCTX) * HDIM;
    const float scale = 0.125f;                      // 1/sqrt(64)

    // --- pass 1: scores ---------------------------------------------------
    for (int t = tid; t <= len; t += 256) {
        const float* kv = (t == len) ? kn_s : (kbase + (size_t)t * HDIM);
        if (t + 256 < len) __builtin_prefetch(kbase + (size_t)(t + 256) * HDIM, 0, 0);
        const float4* k4 = reinterpret_cast<const float4*>(kv);
        const float4* q4 = reinterpret_cast<const float4*>(q_s);
        float s = 0.0f;
#pragma unroll
        for (int i = 0; i < HDIM / 4; ++i) {
            float4 kk = k4[i], qq = q4[i];
            s += kk.x * qq.x + kk.y * qq.y + kk.z * qq.z + kk.w * qq.w;
        }
        sc[t] = s * scale;
    }
    __syncthreads();

    // --- max reduce -------------------------------------------------------
    float m = -1e30f;
    for (int t = tid; t <= len; t += 256) m = fmaxf(m, sc[t]);
#pragma unroll
    for (int o = 16; o > 0; o >>= 1) m = fmaxf(m, __shfl_xor(m, o, 32));
    if ((tid & 31) == 0) red[tid >> 5] = m;
    __syncthreads();
    if (tid == 0) {
        float mm = red[0];
        for (int i = 1; i < 8; ++i) mm = fmaxf(mm, red[i]);
        bc[0] = mm;
    }
    __syncthreads();
    m = bc[0];

    // --- exp + sum --------------------------------------------------------
    float ssum = 0.0f;
    for (int t = tid; t <= len; t += 256) {
        float e = __expf(sc[t] - m);
        sc[t] = e;
        ssum += e;
    }
#pragma unroll
    for (int o = 16; o > 0; o >>= 1) ssum += __shfl_xor(ssum, o, 32);
    __syncthreads();
    if ((tid & 31) == 0) red[tid >> 5] = ssum;
    __syncthreads();
    if (tid == 0) {
        float t2 = 0.0f;
        for (int i = 0; i < 8; ++i) t2 += red[i];
        bc[1] = 1.0f / t2;
    }
    __syncthreads();
    const float inv = bc[1];

    // --- p @ V ------------------------------------------------------------
    const int d = tid & (HDIM - 1);
    const int chunk = tid >> 6;                      // 0..3
    float a = 0.0f;
    for (int t = chunk; t <= len; t += 4) {
        float vv = (t == len) ? vn_s[d] : vbase[(size_t)t * HDIM + d];
        a += sc[t] * vv;
    }
    oacc[chunk][d] = a;
    __syncthreads();
    if (chunk == 0) {
        float o = (oacc[0][d] + oacc[1][d] + oacc[2][d] + oacc[3][d]) * inv;
        out[(size_t)b * DMODEL + h * HDIM + d] = o;
    }
}

// ---------------------------------------------------------------------------
// LayerNorm over the last dim (1024). One block per row.
// ---------------------------------------------------------------------------
__global__ void ln_kernel(const float* __restrict__ xin,
                          const float* __restrict__ g,
                          const float* __restrict__ bta,
                          float* __restrict__ out) {
    const int row = blockIdx.x;
    const int tid = threadIdx.x;
    const float* xr = xin + (size_t)row * DMODEL;

    __shared__ float red[8];
    __shared__ float bc[2];

    float s = 0.0f;
    for (int i = tid; i < DMODEL; i += 256) s += xr[i];
#pragma unroll
    for (int o = 16; o > 0; o >>= 1) s += __shfl_xor(s, o, 32);
    if ((tid & 31) == 0) red[tid >> 5] = s;
    __syncthreads();
    if (tid == 0) {
        float t = 0.0f;
        for (int i = 0; i < 8; ++i) t += red[i];
        bc[0] = t * (1.0f / DMODEL);
    }
    __syncthreads();
    const float mu = bc[0];

    float v = 0.0f;
    for (int i = tid; i < DMODEL; i += 256) {
        float dd = xr[i] - mu;
        v += dd * dd;
    }
#pragma unroll
    for (int o = 16; o > 0; o >>= 1) v += __shfl_xor(v, o, 32);
    if ((tid & 31) == 0) red[tid >> 5] = v;
    __syncthreads();
    if (tid == 0) {
        float t = 0.0f;
        for (int i = 0; i < 8; ++i) t += red[i];
        bc[1] = rsqrtf(t * (1.0f / DMODEL) + 1e-5f);
    }
    __syncthreads();
    const float r = bc[1];

    for (int i = tid; i < DMODEL; i += 256)
        out[(size_t)row * DMODEL + i] = (xr[i] - mu) * r * g[i] + bta[i];
}

// ---------------------------------------------------------------------------
// Launch
// ---------------------------------------------------------------------------
extern "C" void kernel_launch(void* const* d_in, const int* in_sizes, int n_in,
                              void* d_out, int out_size, void* d_ws, size_t ws_size,
                              hipStream_t stream) {
    const float* x        = (const float*)d_in[0];
    const float* k_cache  = (const float*)d_in[1];
    const float* v_cache  = (const float*)d_in[2];
    /* d_in[3] attn_mask  : implied by kv_cache_len, unused */
    const int*   kv_len   = (const int*)  d_in[4];
    /* d_in[5] batch_idx  : arange, unused */
    const float* qkv_w    = (const float*)d_in[6];
    const float* qkv_b    = (const float*)d_in[7];
    const float* out_w    = (const float*)d_in[8];
    const float* out_b    = (const float*)d_in[9];
    const float* ln1_w    = (const float*)d_in[10];
    const float* ln1_b    = (const float*)d_in[11];
    const float* ln2_w    = (const float*)d_in[12];
    const float* ln2_b    = (const float*)d_in[13];
    const float* mlp_w1   = (const float*)d_in[14];
    const float* mlp_b1   = (const float*)d_in[15];
    const float* mlp_w2   = (const float*)d_in[16];
    const float* mlp_b2   = (const float*)d_in[17];

    float* ws   = (float*)d_ws;
    float* qkv  = ws;                         // 64*3072
    float* attn = qkv  + BATCH * 3 * DMODEL;  // 64*1024
    float* x1   = attn + BATCH * DMODEL;      // 64*1024
    float* x2   = x1   + BATCH * DMODEL;      // 64*1024
    float* hbuf = x2   + BATCH * DMODEL;      // 64*4096
    float* out  = (float*)d_out;

    // 1) qkv = x @ qkv_w + qkv_b               (64 x 1024 x 3072)
    gemm64_wmma<<<3 * DMODEL / 64, 256, 0, stream>>>(
        x, qkv_w, qkv_b, nullptr, qkv, DMODEL, 3 * DMODEL, 0);

    // 2) attention (reads caches; fresh k/v virtually placed at t == len)
    attn_kernel<<<dim3(BATCH, NHEAD), 256, 0, stream>>>(
        qkv, k_cache, v_cache, kv_len, attn);

    // 3) x1 = x + attn @ out_w + out_b         (64 x 1024 x 1024)
    gemm64_wmma<<<DMODEL / 64, 256, 0, stream>>>(
        attn, out_w, out_b, x, x1, DMODEL, DMODEL, 0);

    // 4) x2 = LN1(x1)
    ln_kernel<<<BATCH, 256, 0, stream>>>(x1, ln1_w, ln1_b, x2);

    // 5) h = relu(x2 @ mlp_w1 + mlp_b1)        (64 x 1024 x 4096)
    gemm64_wmma<<<DFF_ / 64, 256, 0, stream>>>(
        x2, mlp_w1, mlp_b1, nullptr, hbuf, DMODEL, DFF_, 1);

    // 6) x1 = x2 + h @ mlp_w2 + mlp_b2         (64 x 4096 x 1024)
    gemm64_wmma<<<DMODEL / 64, 256, 0, stream>>>(
        hbuf, mlp_w2, mlp_b2, x2, x1, DFF_, DMODEL, 0);

    // 7) out = LN2(x1)
    ln_kernel<<<BATCH, 256, 0, stream>>>(x1, ln2_w, ln2_b, out);
}
